// Encoder_15530601742980
// MI455X (gfx1250) — matrix operand order repaired and checked by reading, not compiled, not run on hardware
//
#include <hip/hip_runtime.h>

// ---------------------------------------------------------------------------
// Bidirectional 2-layer GRU encoder for MI455X (gfx1250, wave32, WMMA).
// bf16 WMMA (v_wmma_f32_16x16x32_bf16), f32 accumulate + f32 gate math.
// Weights pre-swizzled into B-fragment layout -> 32B/lane contiguous loads.
// ---------------------------------------------------------------------------

typedef __attribute__((ext_vector_type(16))) __bf16 v16bf;
typedef __attribute__((ext_vector_type(8)))  float  v8f;

#define TT   128   // sequence length
#define BB   64    // batch
#define EMBD 512   // embedding dim
#define UU   512   // hidden units
#define TH3  1536  // 3 * UU (z|r|n gate columns)
#define KT   16    // K tiles per matrix (512/32)
#define CT   96    // column tiles per matrix (1536/16)

__device__ __forceinline__ unsigned short f2bfbits(float f) {
  unsigned int u = __float_as_uint(f);
  u += 0x7fffu + ((u >> 16) & 1u);   // round-to-nearest-even
  return (unsigned short)(u >> 16);
}

__device__ __forceinline__ float sigf(float x) {
  return 1.0f / (1.0f + __expf(-x));
}

// A fragment: 16 rows x 32 K from a row-major bf16 matrix (ld elements).
// Lane layout: lanes 0-15 row M=lane, K {kg*8+0..7, 16+kg*8+0..7} with kg=0;
// lanes 16-31 same rows with kg=1. Each half is 8 contiguous bf16 = 16 bytes.
__device__ __forceinline__ v16bf load_tileA(const unsigned short* __restrict__ base,
                                            int ld, int row0, int k0, int lane) {
  const int m  = lane & 15;
  const int kg = lane >> 4;
  const unsigned short* rp = base + (size_t)(row0 + m) * ld + k0 + (kg << 3);
  v16bf a;
  __builtin_memcpy(&a, rp, 16);                         // K run 0..7
  __builtin_memcpy((char*)&a + 16, rp + 16, 16);        // K run 16..23
  return a;
}

// B fragment from pre-swizzled weights: tile (ktile, ctile) stored as
// [lane][16 bf16] contiguous -> one 32-byte load per lane.
__device__ __forceinline__ v16bf load_tileB_sw(const unsigned short* __restrict__ base,
                                               int ktile, int ctile, int lane) {
  const unsigned short* p =
      base + (((size_t)(ktile * CT + ctile) << 5) + lane) * 16;
  v16bf b;
  __builtin_memcpy(&b, p, 32);
  return b;
}

__device__ __forceinline__ v8f wmma_bf16(v16bf a, v16bf b, v8f c) {
  return __builtin_amdgcn_wmma_f32_16x16x32_bf16(
      /*neg_a=*/false, a, /*neg_b=*/false, b,
      /*c_mod=*/(short)0, c, /*reuse_a=*/false, /*reuse_b=*/false);
}

// ---------------------------------------------------------------------------
// Prep kernels
// ---------------------------------------------------------------------------

// Swizzled weight layout per matrix: [ktile(16)][ctile(96)][lane(32)][v(16)].
// Lane holds column ctile*16 + (lane&15); values v are K = ktile*32 +
// (lane>>4)*8 + (v&7) + (v>>3)*16 (matches A-fragment K order).
__global__ __launch_bounds__(256) void conv_weights(
    const float* __restrict__ Wf, const float* __restrict__ Uf,
    const float* __restrict__ Wb, const float* __restrict__ Ub,
    unsigned short* __restrict__ wbf) {
  const size_t MATSZ = (size_t)UU * TH3;
  size_t idx = (size_t)blockIdx.x * 256 + threadIdx.x;
  if (idx >= 8 * MATSZ) return;
  size_t e = idx % MATSZ;              // element within swizzled matrix
  int r = (int)(idx / MATSZ);          // [dir][layer][mat]
  int mat = r & 1, layer = (r >> 1) & 1, dir = r >> 2;

  int v     = (int)(e & 15);
  int lane  = (int)((e >> 4) & 31);
  int tile  = (int)(e >> 9);           // 0..1535
  int ctile = tile % CT;
  int ktile = tile / CT;
  int k   = ktile * 32 + ((lane >> 4) << 3) + (v & 7) + ((v >> 3) << 4);
  int col = ctile * 16 + (lane & 15);

  const float* src = dir ? (mat ? Ub : Wb) : (mat ? Uf : Wf);
  wbf[idx] = f2bfbits(src[(size_t)layer * MATSZ + (size_t)k * TH3 + col]);
}

// xb layout: [t][b][e] bf16 (row-major; A fragments load 2x16B runs)
__global__ __launch_bounds__(256) void embed_gather(
    const int* __restrict__ tokens, const float* __restrict__ emb,
    unsigned short* __restrict__ xb) {
  size_t idx = (size_t)blockIdx.x * 256 + threadIdx.x;
  if (idx >= (size_t)TT * BB * EMBD) return;
  int e = (int)(idx & (EMBD - 1));
  int b = (int)((idx >> 9) & (BB - 1));
  int t = (int)(idx >> 15);
  int tok = tokens[b * TT + t];
  xb[idx] = f2bfbits(emb[(size_t)tok * EMBD + e]);
}

// h layout: [buf(2)][dir][layer][b][u]; init into buffer 0
__global__ __launch_bounds__(256) void init_h(
    const float* __restrict__ sf, const float* __restrict__ sb,
    float* __restrict__ h32, unsigned short* __restrict__ h16) {
  size_t idx = (size_t)blockIdx.x * 256 + threadIdx.x;
  if (idx >= (size_t)2 * 2 * BB * UU) return;
  int u = (int)(idx & (UU - 1));
  int b = (int)((idx >> 9) & (BB - 1));
  int layer = (int)((idx >> 15) & 1);
  int dir = (int)(idx >> 16);
  float v = (dir ? sb : sf)[(size_t)layer * BB * UU + (size_t)b * UU + u];
  size_t o = ((size_t)dir * 2 + layer) * (size_t)BB * UU + (size_t)b * UU + u;
  h32[o] = v;
  h16[o] = f2bfbits(v);
}

// ---------------------------------------------------------------------------
// Per-timestep GRU layer update (both directions in one launch).
// 256 waves: dir(2) x m-tile(4) x u-tile(32); one 16x16 output tile per wave,
// 6 f32 WMMA accumulators (gi_z,gi_r,gi_n,gh_z,gh_r,gh_n), K=512 in 16 steps.
// Double-buffered h: read parity P = s&1, write Q = P^1.
// ---------------------------------------------------------------------------
template <int LAYER>
__global__ __launch_bounds__(256) void gru_step(
    const unsigned short* __restrict__ xb,
    const unsigned short* __restrict__ wbf,
    const float* __restrict__ bias_f,
    const float* __restrict__ bias_b,
    const int* __restrict__ tokens,
    float* __restrict__ h32,
    unsigned short* __restrict__ h16,
    float* __restrict__ out,
    int s) {
  const int P = s & 1, Q = P ^ 1;
  const int lane = threadIdx.x & 31;
  const int wid  = (blockIdx.x * 256 + threadIdx.x) >> 5;  // 0..255
  const int dir  = wid >> 7;
  const int rem  = wid & 127;
  const int row0 = (rem >> 5) << 4;   // batch tile: 0,16,32,48
  const int ut   = rem & 31;          // unit tile index 0..31
  const int col0 = ut << 4;
  const int t    = dir ? (TT - 1 - s) : s;

  const size_t HSZ   = (size_t)BB * UU;
  const size_t MATSZ = (size_t)UU * TH3;

  const unsigned short* Wm = wbf + (((size_t)dir * 2 + LAYER) * 2 + 0) * MATSZ;
  const unsigned short* Um = wbf + (((size_t)dir * 2 + LAYER) * 2 + 1) * MATSZ;
  // gi input: layer0 <- embeddings at t; layer1 <- layer0's *new* h (buffer Q)
  const unsigned short* Ax = (LAYER == 0)
      ? (xb + (size_t)t * BB * EMBD)
      : (h16 + (((size_t)Q * 2 + dir) * 2 + 0) * HSZ);
  const unsigned short* Ah = h16 + (((size_t)P * 2 + dir) * 2 + LAYER) * HSZ;

  v8f giz = {}, gir = {}, gin = {}, ghz = {}, ghr = {}, ghn = {};
  for (int kt = 0; kt < KT; ++kt) {
    const int k = kt << 5;
    if (kt + 1 < KT) {
      __builtin_prefetch(Ax + (size_t)row0 * 512 + k + 32, 0, 0);
      __builtin_prefetch(Ah + (size_t)row0 * 512 + k + 32, 0, 0);
    }
    v16bf a1 = load_tileA(Ax, 512, row0, k, lane);
    v16bf a2 = load_tileA(Ah, 512, row0, k, lane);
    v16bf bz = load_tileB_sw(Wm, kt, ut,      lane);
    v16bf br = load_tileB_sw(Wm, kt, ut + 32, lane);
    v16bf bn = load_tileB_sw(Wm, kt, ut + 64, lane);
    giz = wmma_bf16(a1, bz, giz);
    gir = wmma_bf16(a1, br, gir);
    gin = wmma_bf16(a1, bn, gin);
    bz = load_tileB_sw(Um, kt, ut,      lane);
    br = load_tileB_sw(Um, kt, ut + 32, lane);
    bn = load_tileB_sw(Um, kt, ut + 64, lane);
    ghz = wmma_bf16(a2, bz, ghz);
    ghr = wmma_bf16(a2, br, ghr);
    ghn = wmma_bf16(a2, bn, ghn);
  }

  const float* bgi = (dir ? bias_b : bias_f) + (size_t)LAYER * 2 * TH3;
  const float* bgh = bgi + TH3;
  const float* hp32    = h32 + (((size_t)P * 2 + dir) * 2 + LAYER) * HSZ;
  float*       ho32    = h32 + (((size_t)Q * 2 + dir) * 2 + LAYER) * HSZ;
  unsigned short* ho16 = h16 + (((size_t)Q * 2 + dir) * 2 + LAYER) * HSZ;

  const int n  = lane & 15;
  const int kg = lane >> 4;
  const int cu = col0 + n;
  const float bz0 = bgi[cu]       + bgh[cu];
  const float br0 = bgi[cu + 512] + bgh[cu + 512];
  const float bgn = bgi[cu + 1024];
  const float bhn = bgh[cu + 1024];

#pragma unroll
  for (int j = 0; j < 8; ++j) {
    const int b = row0 + j + (kg << 3);       // C/D layout: VGPR j -> M=j / j+8
    const float z  = sigf(giz[j] + ghz[j] + bz0);
    const float r  = sigf(gir[j] + ghr[j] + br0);
    const float nn = tanhf(gin[j] + bgn + r * (ghn[j] + bhn));
    const float hp = hp32[(size_t)b * UU + cu];
    float hv = z * hp + (1.0f - z) * nn;
    if (tokens[b * TT + t] == 0) hv = hp;     // mask: carry state
    ho32[(size_t)b * UU + cu] = hv;
    ho16[(size_t)b * UU + cu] = f2bfbits(hv);
    if (LAYER == 1)
      out[(size_t)b * (TT * 1024) + (size_t)t * 1024 + (size_t)dir * 512 + cu] = hv;
  }
}

// state tail: concat(hf[-1], hb[-1]); after s=127, Q = 0 -> final h in buf 0.
__global__ __launch_bounds__(256) void final_state(
    const float* __restrict__ h32, float* __restrict__ out) {
  size_t idx = (size_t)blockIdx.x * 256 + threadIdx.x;
  if (idx >= (size_t)2 * BB * UU) return;
  int u = (int)(idx & (UU - 1));
  int b = (int)((idx >> 9) & (BB - 1));
  int dir = (int)(idx >> 15);
  float v = h32[((size_t)dir * 2 + 1) * (size_t)BB * UU + (size_t)b * UU + u];
  out[(size_t)BB * TT * 1024 + (size_t)b * 1024 + (size_t)dir * 512 + u] = v;
}

// ---------------------------------------------------------------------------
// Host launcher
// ---------------------------------------------------------------------------
extern "C" void kernel_launch(void* const* d_in, const int* in_sizes, int n_in,
                              void* d_out, int out_size, void* d_ws, size_t ws_size,
                              hipStream_t stream) {
  (void)in_sizes; (void)n_in; (void)out_size; (void)ws_size;
  const int*   tokens  = (const int*)d_in[0];
  const float* state_f = (const float*)d_in[1];
  const float* state_b = (const float*)d_in[2];
  const float* emb     = (const float*)d_in[3];
  const float* Wf      = (const float*)d_in[4];
  const float* Uf      = (const float*)d_in[5];
  const float* bf      = (const float*)d_in[6];
  const float* Wb      = (const float*)d_in[7];
  const float* Ub      = (const float*)d_in[8];
  const float* bb      = (const float*)d_in[9];
  float* out = (float*)d_out;

  // Workspace layout (bytes):
  //   xb  : [T][B][512] bf16                          =  8,388,608
  //   wbf : [dir][layer][W|U] swizzled fragments bf16 = 12,582,912
  //   h32 : [buf2][dir][layer][64][512] f32           =  1,048,576
  //   h16 : same, bf16                                =    524,288
  char* ws = (char*)d_ws;
  unsigned short* xb  = (unsigned short*)(ws);
  unsigned short* wbf = (unsigned short*)(ws + 8388608);
  float*          h32 = (float*)(ws + 20971520);
  unsigned short* h16 = (unsigned short*)(ws + 23068672);

  conv_weights<<<24576, 256, 0, stream>>>(Wf, Uf, Wb, Ub, wbf);
  embed_gather<<<16384, 256, 0, stream>>>(tokens, emb, xb);
  init_h<<<512, 256, 0, stream>>>(state_f, state_b, h32, h16);

  for (int s = 0; s < TT; ++s) {
    gru_step<0><<<32, 256, 0, stream>>>(xb, wbf, bf, bb, tokens, h32, h16, out, s);
    gru_step<1><<<32, 256, 0, stream>>>(xb, wbf, bf, bb, tokens, h32, h16, out, s);
  }

  final_state<<<256, 256, 0, stream>>>(h32, out);
}
